// Model_39676907886321
// MI455X (gfx1250) — matrix operand order repaired and checked by reading, not compiled, not run on hardware
//
#include <hip/hip_runtime.h>

typedef __attribute__((ext_vector_type(16))) _Float16 v16h;
typedef __attribute__((ext_vector_type(8)))  _Float16 v8h;
typedef __attribute__((ext_vector_type(2)))  _Float16 v2h;
typedef __attribute__((ext_vector_type(8)))  float    v8f;
typedef __attribute__((ext_vector_type(4)))  float    vf4;

#define B_    4
#define H_    16
#define S_    2048
#define D_    64
#define QTILE 128   // query rows per workgroup: 8 waves x 16
#define KTILE 64    // keys per main-loop iteration
#define NWAVES 8

__device__ __forceinline__ v8f wmma_f16(v16h a, v16h b, v8f c) {
  // D = A(16x32 f16) * B(32x16 f16) + C(16x16 f32)
  return __builtin_amdgcn_wmma_f32_16x16x32_f16(
      /*neg_a=*/false, a, /*neg_b=*/false, b,
      /*c_mod=*/(short)0, c, /*reuse_a=*/false, /*reuse_b=*/false);
}

__launch_bounds__(256)
__global__ void fa_fwd_wmma_f16(const float* __restrict__ Q,
                                const float* __restrict__ K,
                                const float* __restrict__ V,
                                float* __restrict__ O) {
  // K block row-major [k][d] (8 KB), V block transposed [d][k] (8 KB)
  __shared__ __align__(32) _Float16 Klds[KTILE * D_];
  __shared__ __align__(32) _Float16 Vt[D_ * KTILE];

  const int tid  = threadIdx.x;
  const int lane = tid & 31;
  const int wave = tid >> 5;
  const int half = (lane < 16) ? 0 : 1;
  const int lrow = lane & 15;
  const int klo  = half * 16;   // unused; B-fragments use K offset half*8 (lanes 16-31 -> K 8..15, 24..31)

  const int bh = blockIdx.x / (S_ / QTILE);
  const int qb = blockIdx.x % (S_ / QTILE);
  const size_t base = (size_t)bh * S_ * D_;
  const float* Qb = Q + base;
  const float* Kb = K + base;
  const float* Vb = V + base;
  float*       Ob = O + base;

  const int qrow0 = qb * QTILE + wave * 16;

  // ---- Resident B-fragments of Q^T (per lane: query lrow, d = 8h..8h+7, 16+8h..16+8h+7),
  //      softmax scale 1/8 folded in.
  v16h bQ0, bQ1;
  {
    const float* qrow = Qb + (size_t)(qrow0 + lrow) * D_;
    #pragma unroll
    for (int ck = 0; ck < 2; ++ck) {
      const vf4* s = (const vf4*)(qrow + ck * 32 + half * 8);
      vf4 x0 = s[0], x1 = s[1], x2 = s[4], x3 = s[5];
      v16h b;
      #pragma unroll
      for (int i = 0; i < 4; ++i) {
        b[i]      = (_Float16)(x0[i] * 0.125f);
        b[4 + i]  = (_Float16)(x1[i] * 0.125f);
        b[8 + i]  = (_Float16)(x2[i] * 0.125f);
        b[12 + i] = (_Float16)(x3[i] * 0.125f);
      }
      if (ck == 0) bQ0 = b; else bQ1 = b;
    }
  }

  // Per-lane online-softmax state for query row (qrow0 + lrow)
  float m = -1e30f, l = 0.0f;
  v8f acc[4];
  #pragma unroll
  for (int t = 0; t < 4; ++t)
    #pragma unroll
    for (int v = 0; v < 8; ++v) acc[t][v] = 0.0f;

  // Cooperative staging indices (64x64 block, 256 threads)
  const int kr = tid >> 2;           // K row 0..63
  const int kc = (tid & 3) * 16;     // K col 0,16,32,48
  const int vr = (tid >> 3) * 2;     // V row pair 0,2,..,62
  const int vc = (tid & 7) * 8;      // V col 0,8,..,56

  // Register double-buffer for the next block, already converted to f16:
  // K: 16 values (2 x v8h), V: 8 packed key-pairs (8 x v2h)
  v8h hk0, hk1;
  v2h hv[8];

  auto gload = [&](int kb) {
    const vf4* s = (const vf4*)(Kb + (size_t)(kb + kr) * D_ + kc);
    vf4 x0 = s[0], x1 = s[1], x2 = s[2], x3 = s[3];
    #pragma unroll
    for (int i = 0; i < 4; ++i) {
      hk0[i]     = (_Float16)x0[i];
      hk0[4 + i] = (_Float16)x1[i];
      hk1[i]     = (_Float16)x2[i];
      hk1[4 + i] = (_Float16)x3[i];
    }
    const vf4* s0 = (const vf4*)(Vb + (size_t)(kb + vr) * D_ + vc);
    const vf4* s1 = (const vf4*)(Vb + (size_t)(kb + vr + 1) * D_ + vc);
    vf4 a0 = s0[0], a1 = s0[1], b0 = s1[0], b1 = s1[1];
    #pragma unroll
    for (int i = 0; i < 4; ++i) {
      hv[i][0]     = (_Float16)a0[i];
      hv[i][1]     = (_Float16)b0[i];
      hv[4 + i][0] = (_Float16)a1[i];
      hv[4 + i][1] = (_Float16)b1[i];
    }
  };

  gload(0);  // prologue: fill the register buffer with block 0

  for (int kb = 0; kb < S_; kb += KTILE) {
    __syncthreads();  // previous iteration's LDS reads complete

    // ---- Drain register buffer into LDS
    *(v8h*)&Klds[kr * D_ + kc]     = hk0;
    *(v8h*)&Klds[kr * D_ + kc + 8] = hk1;
    #pragma unroll
    for (int i = 0; i < 4; ++i) {
      *(v2h*)&Vt[(vc + i) * KTILE + vr]     = hv[i];
      *(v2h*)&Vt[(vc + 4 + i) * KTILE + vr] = hv[4 + i];
    }
    __syncthreads();

    // ---- Pipeline: issue next block's global loads now; their latency is
    //      hidden under the 16 WMMAs + softmax below. L2-warm two ahead.
    if (kb + 2 * KTILE < S_) {
      __builtin_prefetch(Kb + (size_t)(kb + 2 * KTILE + kr) * D_ + kc, 0, 0);
      __builtin_prefetch(Vb + (size_t)(kb + 2 * KTILE + vr) * D_ + vc, 0, 0);
    }
    if (kb + KTILE < S_) gload(kb + KTILE);

    // ---- S^T = K (Q/8)^T : four 16(key)x16(query) f32 tiles.
    // C-layout: lane = query, VGPR v = key (v + 8*half within tile).
    v8f c[4];
    #pragma unroll
    for (int kt = 0; kt < 4; ++kt) {
      const int krow = kt * 16 + lrow;
      const int kb8  = half * 8;
      v8h lo0 = *(const v8h*)&Klds[krow * D_ + kb8];
      v8h hi0 = *(const v8h*)&Klds[krow * D_ + kb8 + 16];
      v8h lo1 = *(const v8h*)&Klds[krow * D_ + 32 + kb8];
      v8h hi1 = *(const v8h*)&Klds[krow * D_ + 32 + kb8 + 16];
      v16h aK0, aK1;
      #pragma unroll
      for (int i = 0; i < 8; ++i) {
        aK0[i] = lo0[i]; aK0[8 + i] = hi0[i];
        aK1[i] = lo1[i]; aK1[8 + i] = hi1[i];
      }
      v8f z;
      #pragma unroll
      for (int v = 0; v < 8; ++v) z[v] = 0.0f;
      z     = wmma_f16(aK0, bQ0, z);
      c[kt] = wmma_f16(aK1, bQ1, z);
    }

    // ---- Online softmax: in-lane folds + ONE cross-half shuffle each
    float rm = c[0][0];
    #pragma unroll
    for (int kt = 0; kt < 4; ++kt)
      #pragma unroll
      for (int v = 0; v < 8; ++v) rm = fmaxf(rm, c[kt][v]);
    rm = fmaxf(rm, __shfl_xor(rm, 16, 32));

    const float mn    = fmaxf(m, rm);
    const float alpha = __expf(m - mn);
    m = mn;

    float rs = 0.0f;
    #pragma unroll
    for (int kt = 0; kt < 4; ++kt)
      #pragma unroll
      for (int v = 0; v < 8; ++v) {
        const float p = __expf(c[kt][v] - mn);
        c[kt][v] = p;
        rs += p;
      }
    rs += __shfl_xor(rs, 16, 32);
    l = fmaf(l, alpha, rs);

    // Broadcast alpha (lane-indexed by query) into acc's row-across-VGPR form
    float af[8];
    #pragma unroll
    for (int v = 0; v < 8; ++v) af[v] = __shfl(alpha, 8 * half + v, 32);
    #pragma unroll
    for (int t = 0; t < 4; ++t)
      #pragma unroll
      for (int v = 0; v < 8; ++v) acc[t][v] *= af[v];

    // ---- P A-fragments directly from S^T fragments (pure in-lane cvt):
    // aP0 covers keys 0..31, aP1 covers keys 32..63.
    v16h aP0, aP1;
    #pragma unroll
    for (int v = 0; v < 8; ++v) {
      aP0[v]     = (_Float16)c[0][v];
      aP0[8 + v] = (_Float16)c[1][v];
      aP1[v]     = (_Float16)c[2][v];
      aP1[8 + v] = (_Float16)c[3][v];
    }

    // ---- O += P * V : 4 output d-tiles, K = 64 in two chunks
    #pragma unroll
    for (int t = 0; t < 4; ++t) {
      const v8h* vp = (const v8h*)&Vt[(t * 16 + lrow) * KTILE + half * 8];
      const v16h bv0 = __builtin_shufflevector(vp[0], vp[2], 0, 1, 2, 3, 4, 5, 6, 7, 8, 9, 10, 11, 12, 13, 14, 15);
      const v16h bv1 = __builtin_shufflevector(vp[4], vp[6], 0, 1, 2, 3, 4, 5, 6, 7, 8, 9, 10, 11, 12, 13, 14, 15);
      acc[t] = wmma_f16(aP0, bv0, acc[t]);
      acc[t] = wmma_f16(aP1, bv1, acc[t]);
    }
  }

  // ---- Epilogue: bring per-lane row sums into row-across-VGPR form, store
  float lr[8];
  #pragma unroll
  for (int v = 0; v < 8; ++v) lr[v] = __shfl(l, 8 * half + v, 32);
  #pragma unroll
  for (int v = 0; v < 8; ++v) {
    const float invl = 1.0f / lr[v];
    const int row = qrow0 + v + 8 * half;
    #pragma unroll
    for (int t = 0; t < 4; ++t) {
      Ob[(size_t)row * D_ + t * 16 + lrow] = acc[t][v] * invl;
    }
  }
}

extern "C" void kernel_launch(void* const* d_in, const int* in_sizes, int n_in,
                              void* d_out, int out_size, void* d_ws, size_t ws_size,
                              hipStream_t stream) {
  (void)in_sizes; (void)n_in; (void)out_size; (void)d_ws; (void)ws_size;
  const float* q = (const float*)d_in[0];
  const float* k = (const float*)d_in[1];
  const float* v = (const float*)d_in[2];
  float* o = (float*)d_out;
  dim3 grid(B_ * H_ * (S_ / QTILE));
  dim3 block(256);
  fa_fwd_wmma_f16<<<grid, block, 0, stream>>>(q, k, v, o);
}
